// OneD_RNN_wavefxn_60687887893250
// MI455X (gfx1250) — compile-verified, matrix operand order checked by reading, not run on hardware
//
#include <hip/hip_runtime.h>
#include <hip/hip_bf16.h>
#include <math.h>

#define NSITES 144
#define NH 256
#define G3 768        // 3*NH
#define BATCH 1024
#define BT 16         // batch tile (rows) per workgroup
#define TPB 512       // 16 waves of 32
#define NT 3          // n-tiles per wave: 48 tiles / 16 waves
#define KF 8          // k-fragments: 256 / 32

typedef __bf16 bf16;
typedef __attribute__((ext_vector_type(16))) __bf16 v16bf;
typedef __attribute__((ext_vector_type(8)))  float  v8f;

__device__ __forceinline__ unsigned mix32(unsigned x) {
    x ^= x >> 16; x *= 0x85EBCA6Bu;
    x ^= x >> 13; x *= 0xC2B2AE35u;
    x ^= x >> 16; return x;
}

__global__ __launch_bounds__(TPB, 1)
void gru_wavefxn_sampler(const float* __restrict__ kernel,      // [2, 768]
                         const float* __restrict__ rec_kernel,  // [256, 768]
                         const float* __restrict__ bias,        // [2, 768]
                         const float* __restrict__ dense_w,     // [256, 2]
                         const float* __restrict__ dense_b,     // [2]
                         float* __restrict__ out_samples,       // [1024, 144]
                         float* __restrict__ out_logp)          // [1024]
{
    __shared__ float GH[BT * G3];                       // 48KB: gh = h@Wrec (no bias)
    __shared__ float Hst[BT * NH];                      // 16KB: hidden state f32
    __shared__ __align__(32) bf16 Apack[KF * 512];      //  8KB: h packed as WMMA A frags
    __shared__ float Kmat[2 * G3];                      //  6KB: input kernel rows
    __shared__ float B0[G3];                            //  3KB: bias[0]
    __shared__ float B1[G3];                            //  3KB: bias[1] (folded in gate phase)
    __shared__ float DW[NH * 2];                        //  2KB
    __shared__ float DB[2];
    __shared__ int   Sprev[BT];
    __shared__ float LogP[BT];

    const int tid  = threadIdx.x;
    const int lane = tid & 31;
    const int wave = tid >> 5;
    const int b0   = blockIdx.x * BT;
    const int hi   = lane >> 4;    // which 16-lane half
    const int nlo  = lane & 15;    // N (column) within tile / row within C half

    // ---------------- one-time LDS staging ----------------
    for (int i = tid; i < 2 * G3;  i += TPB) Kmat[i] = kernel[i];
    for (int i = tid; i < G3;      i += TPB) B0[i]   = bias[i];
    for (int i = tid; i < G3;      i += TPB) B1[i]   = bias[G3 + i];
    for (int i = tid; i < NH * 2;  i += TPB) DW[i]   = dense_w[i];
    if (tid < 2)  DB[tid] = dense_b[tid];
    if (tid < BT) { Sprev[tid] = -1; LogP[tid] = 0.0f; }
    for (int i = tid; i < BT * NH;  i += TPB) Hst[i] = 0.0f;
    for (int i = tid; i < KF * 512; i += TPB) Apack[i] = (bf16)0.0f;

    // ------- weights-stationary: rec_kernel bf16 resident in VGPRs -------
    // B-matrix 32x16 bf16 fragment layout (mirrors ISA A layout):
    // lanes 0-15 hold N=lane, K in {0..7,16..23}; lanes 16-31 hold N=lane-16, K in {8..15,24..31}
    v16bf w[NT][KF];
    #pragma unroll
    for (int t = 0; t < NT; ++t) {
        const int cb = (wave * NT + t) * 16;            // global column base
        #pragma unroll
        for (int f = 0; f < KF; ++f) {
            v16bf wf;
            #pragma unroll
            for (int e = 0; e < 16; ++e) {
                const int kk = (e & 7) + 8 * hi + 16 * (e >> 3);
                wf[e] = (bf16)rec_kernel[(f * 32 + kk) * G3 + cb + nlo];
            }
            w[t][f] = wf;
        }
    }
    __syncthreads();

    // ---------------- 144 sequential autoregressive steps ----------------
    #pragma unroll 1
    for (int step = 0; step < NSITES; ++step) {
        // === phase 1: gh = h @ Wrec  (24 WMMAs / wave, zero-init C) ===
        v8f acc[NT];
        #pragma unroll
        for (int t = 0; t < NT; ++t) acc[t] = (v8f)0.0f;
        #pragma unroll
        for (int f = 0; f < KF; ++f) {
            v16bf a = *(const v16bf*)&Apack[f * 512 + lane * 16];
            #pragma unroll
            for (int t = 0; t < NT; ++t)
                acc[t] = __builtin_amdgcn_wmma_f32_16x16x32_bf16(
                    false, a, false, w[t][f], (short)0, acc[t], false, false);
        }
        #pragma unroll
        for (int t = 0; t < NT; ++t) {
            const int cb = (wave * NT + t) * 16;
            #pragma unroll
            for (int r = 0; r < 8; ++r)                 // C: vgpr r = row r (+8 for hi half)
                GH[(r + 8 * hi) * G3 + cb + nlo] = acc[t][r];
        }
        __syncthreads();

        // === phase 2: GRU gates + h update + bf16 A repack ===
        {
            const int m  = tid >> 5;                    // sample row in tile
            const int j0 = (tid & 31) * 8;              // 8 hidden units per thread
            const int sp = Sprev[m];
            #pragma unroll
            for (int jj = 0; jj < 8; ++jj) {
                const int j = j0 + jj;
                float xz = B0[j], xr = B0[256 + j], xh = B0[512 + j];
                if (sp >= 0) {                          // x one-hot -> row select of kernel
                    const float* kr = &Kmat[sp * G3];
                    xz += kr[j]; xr += kr[256 + j]; xh += kr[512 + j];
                }
                const float hz = GH[m * G3 + j]       + B1[j];
                const float hr = GH[m * G3 + 256 + j] + B1[256 + j];
                const float hn = GH[m * G3 + 512 + j] + B1[512 + j];
                const float z  = 1.0f / (1.0f + __expf(-(xz + hz)));
                const float r  = 1.0f / (1.0f + __expf(-(xr + hr)));
                const float hh = tanhf(xh + r * hn);
                const float hnew = z * Hst[m * NH + j] + (1.0f - z) * hh;
                Hst[m * NH + j] = hnew;
                // scatter into WMMA A-fragment layout for next step
                const int kk = j & 31, sl = j >> 5;
                const int ln = m + 16 * ((kk >> 3) & 1);
                const int e  = (kk & 7) + 8 * (kk >> 4);
                Apack[sl * 512 + ln * 16 + e] = (bf16)hnew;
            }
        }
        __syncthreads();

        // === phase 3: dense head + softmax + categorical sample (wave m -> sample m) ===
        {
            const int m = wave;
            float s0 = 0.0f, s1 = 0.0f;
            #pragma unroll
            for (int jj = 0; jj < 8; ++jj) {
                const int j = lane * 8 + jj;
                const float h = Hst[m * NH + j];
                s0 += h * DW[j * 2 + 0];
                s1 += h * DW[j * 2 + 1];
            }
            #pragma unroll
            for (int off = 16; off > 0; off >>= 1) {    // wave32 reduction
                s0 += __shfl_xor(s0, off, 32);
                s1 += __shfl_xor(s1, off, 32);
            }
            if (lane == 0) {
                const float l0 = s0 + DB[0], l1 = s1 + DB[1];
                const float mx = fmaxf(l0, l1);
                const float e0 = __expf(l0 - mx), e1 = __expf(l1 - mx);
                const float inv = 1.0f / (e0 + e1);
                const float p0 = e0 * inv, p1 = e1 * inv;
                const float q0 = (1e-10f + p0) / (2e-10f + p0 + p1);
                // deterministic counter-based uniform (stable across graph replays)
                unsigned x = (unsigned)(step * BATCH + b0 + m) * 0x9E3779B9u + 0x7F4A7C15u;
                const float u = (float)(mix32(x) >> 8) * (1.0f / 16777216.0f);
                const int s = (u < q0) ? 0 : 1;
                LogP[m] += __logf(1e-10f + (s ? p1 : p0));
                Sprev[m] = s;
                out_samples[(size_t)(b0 + m) * NSITES + step] = (float)s;
            }
        }
        __syncthreads();
    }

    if (tid < BT) out_logp[b0 + tid] = LogP[tid];
}

extern "C" void kernel_launch(void* const* d_in, const int* in_sizes, int n_in,
                              void* d_out, int out_size, void* d_ws, size_t ws_size,
                              hipStream_t stream) {
    const float* kern   = (const float*)d_in[0];
    const float* rec    = (const float*)d_in[1];
    const float* bias   = (const float*)d_in[2];
    const float* dw     = (const float*)d_in[3];
    const float* db     = (const float*)d_in[4];
    float* out          = (float*)d_out;
    float* out_samples  = out;                               // 1024*144
    float* out_logp     = out + (size_t)BATCH * NSITES;      // 1024
    gru_wavefxn_sampler<<<dim3(BATCH / BT), TPB, 0, stream>>>(
        kern, rec, bias, dw, db, out_samples, out_logp);
}
